// RelGATv2EncoderTG_31473520345156
// MI455X (gfx1250) — matrix operand order repaired and checked by reading, not compiled, not run on hardware
//
#include <hip/hip_runtime.h>
#include <hip/hip_bf16.h>
#include <math.h>

// Problem constants (match reference)
#define DMODEL   128
#define HEADS    4
#define HD       512      // HEADS * DMODEL
#define RELDIM   64
#define NRELS    64
#define LAYERS   2
#define NEG_SLOPE 0.2f
#define LN_EPS    1e-5f

typedef __attribute__((ext_vector_type(16))) __bf16 v16bf;
typedef __attribute__((ext_vector_type(8)))  float  v8f;

union FragB { v16bf v; uint4 q[2]; };

__device__ __forceinline__ float warp_sum32(float v) {
    #pragma unroll
    for (int m = 16; m >= 1; m >>= 1) v += __shfl_xor(v, m, 32);
    return v;
}

__device__ __forceinline__ float lrelu(float x) {
    return x > 0.0f ? x : x * NEG_SLOPE;
}

// float atomic-max via int/uint bit trick (works with -inf init)
__device__ __forceinline__ void atomicMaxF(float* addr, float val) {
    if (val >= 0.0f) atomicMax((int*)addr, __float_as_int(val));
    else             atomicMin((unsigned int*)addr, __float_as_uint(val));
}

// ---------------------------------------------------------------- utility
__global__ void fill_f32(float* __restrict__ p, float v, long n) {
    long i = (long)blockIdx.x * blockDim.x + threadIdx.x;
    if (i < n) p[i] = v;
}

__global__ void cvt_f32_bf16(const float* __restrict__ in, __bf16* __restrict__ out, long n) {
    long i = (long)blockIdx.x * blockDim.x + threadIdx.x;
    if (i < n) out[i] = (__bf16)in[i];
}

// W [DMODEL][HD] f32  ->  Wt [HD][DMODEL] bf16 (transposed, K contiguous)
__global__ void transpose_w_bf16(const float* __restrict__ W, __bf16* __restrict__ Wt) {
    int i = blockIdx.x * blockDim.x + threadIdx.x;
    if (i >= HD * DMODEL) return;
    int c = i / DMODEL;     // output column of GEMM (0..511)
    int k = i % DMODEL;     // K index (0..127)
    Wt[i] = (__bf16)W[(size_t)k * HD + c];
}

// xe[r][c] = sum_k rel_emb[r][k] * We[k][c]   (64 x 512, tiny)
__global__ void xe_table(const float* __restrict__ rel, const float* __restrict__ We,
                         float* __restrict__ xe) {
    int i = blockIdx.x * blockDim.x + threadIdx.x;
    if (i >= NRELS * HD) return;
    int r = i / HD, c = i % HD;
    float s = 0.0f;
    #pragma unroll 8
    for (int k = 0; k < RELDIM; ++k) s += rel[r * RELDIM + k] * We[(size_t)k * HD + c];
    xe[i] = s;
}

// ---------------------------------------------------------------- GEMM (WMMA)
// C[nrows][HD] = A[nrows][DMODEL](bf16) * Bt^T  + bias, Bt is [HD][DMODEL] bf16.
// Block = 128 threads (4 waves); each wave: 16 rows x 128 cols (8 WMMA tiles), K=128.
__global__ __launch_bounds__(128) void gemm_bf16_wmma(
    const __bf16* __restrict__ A, const __bf16* __restrict__ Bt,
    const float* __restrict__ bias, float* __restrict__ C, int nrows)
{
    const int lane = threadIdx.x & 31;
    const int wave = threadIdx.x >> 5;
    const int row0 = blockIdx.x * 16;
    const int m    = lane & 15;          // M (A) / N (B,C) index within tile
    const int hi8  = (lane & 16) >> 1;   // A: hi lanes hold K+8 ; C: rows +8
    const int klo  = (lane & 16);        // unused; B hi lanes use hi8 (K+8) like A
    const bool full = (row0 + 16 <= nrows);   // N % 16 == 0 -> always true

    int arow = row0 + m; if (arow >= nrows) arow = nrows - 1;

    // A fragments for K = 0..127 (4 chunks of 32), per ISA 16-bit A 16x32 layout:
    // lane m, VGPR0..3 = K(hi8+0..7), VGPR4..7 = K(16+hi8+0..7)  (pairs packed)
    FragB af[4];
    const __bf16* abase = A + (size_t)arow * DMODEL;
    #pragma unroll
    for (int kc = 0; kc < 4; ++kc) {
        const __bf16* ap = abase + kc * 32 + hi8;
        af[kc].q[0] = *(const uint4*)(ap);
        af[kc].q[1] = *(const uint4*)(ap + 16);
    }

    const int colbase = wave * 128;
    #pragma unroll
    for (int n = 0; n < 8; ++n) {
        const int col = colbase + n * 16 + m;
        const __bf16* brow = Bt + (size_t)col * DMODEL;
        v8f acc = {0.f, 0.f, 0.f, 0.f, 0.f, 0.f, 0.f, 0.f};
        #pragma unroll
        for (int kc = 0; kc < 4; ++kc) {
            // B 32x16 layout: lane n=m holds K = hi8+0..7, 16+hi8+0..7 (pairs packed, matches A)
            FragB bfg;
            const __bf16* bp = brow + kc * 32 + hi8;
            bfg.q[0] = *(const uint4*)(bp);
            bfg.q[1] = *(const uint4*)(bp + 16);
            acc = __builtin_amdgcn_wmma_f32_16x16x32_bf16(
                false, af[kc].v, false, bfg.v, (short)0, acc, false, false);
        }
        const float bv = bias[col];
        // C layout: VGPR r -> row (row0 + hi8 + r), fixed col per lane.
        // One base pointer; 8 stores with immediate offsets r*HD*4 (branch-free).
        float* cp = C + (size_t)(row0 + hi8) * HD + col;
        if (full) {
            #pragma unroll
            for (int r = 0; r < 8; ++r) cp[r * HD] = acc[r] + bv;
        } else {
            #pragma unroll
            for (int r = 0; r < 8; ++r)
                if (row0 + hi8 + r < nrows) cp[r * HD] = acc[r] + bv;
        }
    }
}

// ---------------------------------------------------------------- edge pass A
// One wave per edge: logits[e][h] = att[h] . leaky_relu(xl[src] + xr[dst] + xe[rel])
__global__ __launch_bounds__(256) void edge_logits_k(
    const float* __restrict__ xl, const float* __restrict__ xr,
    const float* __restrict__ xe, const float* __restrict__ att,
    const int* __restrict__ src, const int* __restrict__ dst,
    const int* __restrict__ etype, float* __restrict__ logits,
    float* __restrict__ mx, int E)
{
    const int lane = threadIdx.x & 31;
    const int e = blockIdx.x * 8 + (threadIdx.x >> 5);
    if (e >= E) return;
    const int s = src[e], d = dst[e], r = etype[e];
    const float4* pl = (const float4*)(xl + (size_t)s * HD);
    const float4* pr = (const float4*)(xr + (size_t)d * HD);
    const float4* pe = (const float4*)(xe + (size_t)r * HD);
    const float4* pa = (const float4*)att;
    float lg[HEADS];
    #pragma unroll
    for (int hh = 0; hh < HEADS; ++hh) {
        const int idx = hh * 32 + lane;                 // float4 index: 32*4 = 128 ch
        float4 a = pl[idx], b = pr[idx], c = pe[idx], w = pa[idx];
        float p = lrelu(a.x + b.x + c.x) * w.x
                + lrelu(a.y + b.y + c.y) * w.y
                + lrelu(a.z + b.z + c.z) * w.z
                + lrelu(a.w + b.w + c.w) * w.w;
        lg[hh] = warp_sum32(p);
    }
    if (lane == 0) {
        #pragma unroll
        for (int hh = 0; hh < HEADS; ++hh) {
            logits[(size_t)e * HEADS + hh] = lg[hh];
            atomicMaxF(&mx[(size_t)d * HEADS + hh], lg[hh]);
        }
    }
}

// ---------------------------------------------------------------- edge pass B
// exp(logit - max) in place, accumulate denominator
__global__ __launch_bounds__(256) void edge_exp_k(
    float* __restrict__ logits, const float* __restrict__ mx,
    const int* __restrict__ dst, float* __restrict__ den, long n)
{
    long i = (long)blockIdx.x * blockDim.x + threadIdx.x;
    if (i >= n) return;
    int e = (int)(i >> 2), hh = (int)(i & 3);
    int d = dst[e];
    float v = __expf(logits[i] - mx[(size_t)d * HEADS + hh]);
    logits[i] = v;
    unsafeAtomicAdd(&den[(size_t)d * HEADS + hh], v);
}

// ---------------------------------------------------------------- edge pass C
// One wave per edge: acc[dst] += alpha * xl[src]
__global__ __launch_bounds__(256) void edge_scatter_k(
    const float* __restrict__ xl, const float* __restrict__ ex,
    const float* __restrict__ den, const int* __restrict__ src,
    const int* __restrict__ dst, float* __restrict__ acc, int E)
{
    const int lane = threadIdx.x & 31;
    const int e = blockIdx.x * 8 + (threadIdx.x >> 5);
    if (e >= E) return;
    const int s = src[e], d = dst[e];
    float alpha[HEADS];
    #pragma unroll
    for (int hh = 0; hh < HEADS; ++hh)
        alpha[hh] = ex[(size_t)e * HEADS + hh] / den[(size_t)d * HEADS + hh];
    const float4* pl = (const float4*)(xl + (size_t)s * HD);
    #pragma unroll
    for (int hh = 0; hh < HEADS; ++hh) {
        float4 v = pl[hh * 32 + lane];
        float* o = acc + (size_t)d * HD + hh * 128 + lane * 4;
        unsafeAtomicAdd(o + 0, v.x * alpha[hh]);
        unsafeAtomicAdd(o + 1, v.y * alpha[hh]);
        unsafeAtomicAdd(o + 2, v.z * alpha[hh]);
        unsafeAtomicAdd(o + 3, v.w * alpha[hh]);
    }
}

// ---------------------------------------------------------------- node finish
// out = LayerNorm(h_in + gelu(mean_heads(acc) + b_out))
__global__ __launch_bounds__(256) void node_finish_k(
    const float* __restrict__ acc, const float* __restrict__ h_in,
    const float* __restrict__ b_out, const float* __restrict__ ln_g,
    const float* __restrict__ ln_b, float* __restrict__ h_out, int N)
{
    const int lane = threadIdx.x & 31;
    const int n = blockIdx.x * 8 + (threadIdx.x >> 5);
    if (n >= N) return;
    const float4* pa = (const float4*)(acc + (size_t)n * HD);
    float4 s = {0.f, 0.f, 0.f, 0.f};
    #pragma unroll
    for (int hh = 0; hh < HEADS; ++hh) {
        float4 a = pa[hh * 32 + lane];
        s.x += a.x; s.y += a.y; s.z += a.z; s.w += a.w;
    }
    float4 bo = ((const float4*)b_out)[lane];
    float4 x  = { s.x * 0.25f + bo.x, s.y * 0.25f + bo.y,
                  s.z * 0.25f + bo.z, s.w * 0.25f + bo.w };
    // exact GELU
    float4 g;
    g.x = 0.5f * x.x * (1.0f + erff(x.x * 0.70710678118f));
    g.y = 0.5f * x.y * (1.0f + erff(x.y * 0.70710678118f));
    g.z = 0.5f * x.z * (1.0f + erff(x.z * 0.70710678118f));
    g.w = 0.5f * x.w * (1.0f + erff(x.w * 0.70710678118f));
    float4 h0 = ((const float4*)(h_in + (size_t)n * DMODEL))[lane];
    float4 res = { h0.x + g.x, h0.y + g.y, h0.z + g.z, h0.w + g.w };
    float mean = warp_sum32(res.x + res.y + res.z + res.w) * (1.0f / DMODEL);
    float4 dx = { res.x - mean, res.y - mean, res.z - mean, res.w - mean };
    float var = warp_sum32(dx.x*dx.x + dx.y*dx.y + dx.z*dx.z + dx.w*dx.w) * (1.0f / DMODEL);
    float rstd = rsqrtf(var + LN_EPS);
    float4 gg = ((const float4*)ln_g)[lane];
    float4 bb = ((const float4*)ln_b)[lane];
    float4 o = { dx.x * rstd * gg.x + bb.x, dx.y * rstd * gg.y + bb.y,
                 dx.z * rstd * gg.z + bb.z, dx.w * rstd * gg.w + bb.w };
    ((float4*)(h_out + (size_t)n * DMODEL))[lane] = o;
}

// ---------------------------------------------------------------- host
extern "C" void kernel_launch(void* const* d_in, const int* in_sizes, int n_in,
                              void* d_out, int out_size, void* d_ws, size_t ws_size,
                              hipStream_t stream) {
    const float* x_flat   = (const float*)d_in[0];
    const int*   eidx     = (const int*)d_in[1];
    const int*   etype    = (const int*)d_in[2];
    const float* rel_emb  = (const float*)d_in[3];
    const float* Wl       = (const float*)d_in[4];
    const float* bl       = (const float*)d_in[5];
    const float* Wr       = (const float*)d_in[6];
    const float* br       = (const float*)d_in[7];
    const float* We       = (const float*)d_in[8];
    const float* att      = (const float*)d_in[9];
    const float* b_out    = (const float*)d_in[10];
    const float* ln_g     = (const float*)d_in[11];
    const float* ln_b     = (const float*)d_in[12];
    float* out = (float*)d_out;

    const int N = in_sizes[0] / DMODEL;
    const int E = in_sizes[1] / 2;
    const int* src = eidx;
    const int* dst = eidx + E;

    // workspace carve-up (bytes)
    char* ws = (char*)d_ws;
    size_t off = 0;
    float*  XL   = (float*)(ws + off);  off += (size_t)N * HD * 4;        // xl
    float*  XR   = (float*)(ws + off);  off += (size_t)N * HD * 4;        // xr, reused as acc
    float*  HTMP = (float*)(ws + off);  off += (size_t)N * DMODEL * 4;    // inter-layer h
    __bf16* HBF  = (__bf16*)(ws + off); off += (size_t)N * DMODEL * 2;    // bf16 h
    float*  LOG  = (float*)(ws + off);  off += (size_t)E * HEADS * 4;     // logits -> exp
    float*  MX   = (float*)(ws + off);  off += (size_t)N * HEADS * 4;
    float*  DEN  = (float*)(ws + off);  off += (size_t)N * HEADS * 4;
    __bf16* WLT  = (__bf16*)(ws + off); off += (size_t)DMODEL * HD * 2;
    __bf16* WRT  = (__bf16*)(ws + off); off += (size_t)DMODEL * HD * 2;
    float*  XE   = (float*)(ws + off);  off += (size_t)NRELS * HD * 4;
    (void)ws_size; (void)n_in; (void)out_size;

    const long nND = (long)N * DMODEL;
    const long nNH = (long)N * HEADS;
    const long nNHD = (long)N * HD;
    const long nEH = (long)E * HEADS;
    const int eblocks = (E + 7) / 8;
    const int nblocks = (N + 7) / 8;
    const int gemmBlocks = (N + 15) / 16;

    for (int l = 0; l < LAYERS; ++l) {
        const float* h_in  = (l == 0) ? x_flat : HTMP;
        float*       h_out = (l == LAYERS - 1) ? out : HTMP;

        // bf16 activations + transposed bf16 weights + relation table
        cvt_f32_bf16<<<(int)((nND + 255) / 256), 256, 0, stream>>>(h_in, HBF, nND);
        transpose_w_bf16<<<(HD * DMODEL + 255) / 256, 256, 0, stream>>>(Wl + (size_t)l * DMODEL * HD, WLT);
        transpose_w_bf16<<<(HD * DMODEL + 255) / 256, 256, 0, stream>>>(Wr + (size_t)l * DMODEL * HD, WRT);
        xe_table<<<(NRELS * HD + 255) / 256, 256, 0, stream>>>(rel_emb, We + (size_t)l * RELDIM * HD, XE);

        // xl = h@Wl + bl ; xr = h@Wr + br   (WMMA bf16 -> f32)
        gemm_bf16_wmma<<<gemmBlocks, 128, 0, stream>>>(HBF, WLT, bl + (size_t)l * HD, XL, N);
        gemm_bf16_wmma<<<gemmBlocks, 128, 0, stream>>>(HBF, WRT, br + (size_t)l * HD, XR, N);

        // segment softmax init
        fill_f32<<<(int)((nNH + 255) / 256), 256, 0, stream>>>(MX, -INFINITY, nNH);
        fill_f32<<<(int)((nNH + 255) / 256), 256, 0, stream>>>(DEN, 0.0f, nNH);

        // pass A: logits + segment max
        edge_logits_k<<<eblocks, 256, 0, stream>>>(XL, XR, XE, att + (size_t)l * HEADS * DMODEL,
                                                   src, dst, etype, LOG, MX, E);

        // XR is dead now -> reuse as scatter accumulator
        fill_f32<<<(int)((nNHD + 255) / 256), 256, 0, stream>>>(XR, 0.0f, nNHD);

        // pass B: exp + denominator
        edge_exp_k<<<(int)((nEH + 255) / 256), 256, 0, stream>>>(LOG, MX, dst, DEN, nEH);

        // pass C: weighted scatter of messages
        edge_scatter_k<<<eblocks, 256, 0, stream>>>(XL, LOG, DEN, src, dst, XR, E);

        // head-mean + bias + GELU + residual + LayerNorm
        node_finish_k<<<nblocks, 256, 0, stream>>>(XR, h_in, b_out + (size_t)l * DMODEL,
                                                   ln_g + (size_t)l * DMODEL,
                                                   ln_b + (size_t)l * DMODEL, h_out, N);
    }
}
